// S2E_model_35038343201529
// MI455X (gfx1250) — compile-verified, matrix operand order checked by reading, not run on hardware
//
#include <hip/hip_runtime.h>

// ---------------- problem constants ----------------
#define BB 2
#define SS 2048
#define HH 1024
#define FD 3072
#define FL 3072
#define MAXK 819
#define SPAN 30
#define NEGV (-10000.0f)

// ---------------- bf16 helpers (bit-level, storage = ushort) ----------------
__device__ __forceinline__ unsigned short f2bf(float f) {
  unsigned u = __float_as_uint(f);
  unsigned r = (u + 0x7FFFu + ((u >> 16) & 1u)) >> 16;
  return (unsigned short)r;
}
__device__ __forceinline__ float bf2f(unsigned short h) {
  return __uint_as_float(((unsigned)h) << 16);
}

// ---------------- WMMA types ----------------
typedef __attribute__((ext_vector_type(16))) __bf16 v16bf;
typedef __attribute__((ext_vector_type(8)))  float  v8f;
union FragU { v16bf v; uint4 q[2]; };

// ---------------- transpose-cast: out[n][k] = bf16(in[k][n]) ----------------
__global__ __launch_bounds__(256) void tcast_kernel(
    const float* __restrict__ in, unsigned short* __restrict__ out, int K, int N)
{
  __shared__ unsigned short t[32][33];
  const int k0 = blockIdx.y * 32, n0 = blockIdx.x * 32;
  const int tx = threadIdx.x & 31, ty = threadIdx.x >> 5;  // ty 0..7
#pragma unroll
  for (int i = ty; i < 32; i += 8) {
    int k = k0 + i, n = n0 + tx;
    float v = (k < K && n < N) ? in[(size_t)k * N + n] : 0.f;
    t[i][tx] = f2bf(v);
  }
  __syncthreads();
#pragma unroll
  for (int i = ty; i < 32; i += 8) {
    int n = n0 + i, k = k0 + tx;
    if (n < N && k < K) out[(size_t)n * K + k] = t[tx][i];
  }
}

// ---------------- GEMM: C[M,N] = act(A_bf16[M,K] * B_bf16[N,K]^T + bias) ----
// A: bf16 [M,K] row-major.  B: bf16 [N,K] row-major (K contiguous).
// cmode: 0 = write f32, 1 = write bf16, 2 = accumulate into f32
#define Bb_M 128
#define Bb_N 128
#define Bb_K 32
#define LDT  (Bb_K + 8)

__global__ __launch_bounds__(256) void wmma_gemm_kernel(
    const unsigned short* __restrict__ A, int lda, long long sA,
    const unsigned short* __restrict__ B, int ldb, long long sB,
    const float* __restrict__ bias,
    void* __restrict__ Cp, int ldc, long long sC, int cmode,
    int M, int N, int K, int act)
{
  __shared__ unsigned short As[2][Bb_M * LDT];
  __shared__ unsigned short Bs[2][Bb_N * LDT];
  const int tid  = threadIdx.x;
  const int lane = tid & 31;
  const int wave = tid >> 5;
  const int wm = wave & 3;   // 4 wave rows  (32 M each)
  const int wn = wave >> 2;  // 2 wave cols  (64 N each)
  const int m0 = blockIdx.y * Bb_M;
  const int n0 = blockIdx.x * Bb_N;
  const int z  = blockIdx.z;

  const unsigned short* Az = A + (size_t)z * sA;
  const unsigned short* Bz = B + (size_t)z * sB;

  v8f acc[2][4];
#pragma unroll
  for (int i = 0; i < 2; ++i)
#pragma unroll
    for (int j = 0; j < 4; ++j) acc[i][j] = (v8f){0,0,0,0,0,0,0,0};

  const int kb = (lane & 16) ? 8 : 0;     // per-lane-half K sub-chunk base
  const int r  = tid >> 1;                // staged row within tile (0..127)
  const int h  = (tid & 1) << 4;          // K half (0 or 16)
  const bool aIn = (m0 + r) < M;
  const bool bIn = (n0 + r) < N;
  const unsigned short* Aptr = Az + (size_t)(m0 + r) * lda + h;
  const unsigned short* Bptr = Bz + (size_t)(n0 + r) * ldb + h;

  const int ntiles = K / Bb_K;            // K is a multiple of 32 everywhere

  // preload tile 0
  {
    uint4 a0 = {0,0,0,0}, a1 = {0,0,0,0}, b0 = {0,0,0,0}, b1 = {0,0,0,0};
    if (aIn) { const uint4* p = (const uint4*)Aptr; a0 = p[0]; a1 = p[1]; }
    if (bIn) { const uint4* p = (const uint4*)Bptr; b0 = p[0]; b1 = p[1]; }
    uint4* da = (uint4*)(&As[0][r * LDT + h]); da[0] = a0; da[1] = a1;
    uint4* db = (uint4*)(&Bs[0][r * LDT + h]); db[0] = b0; db[1] = b1;
  }
  __syncthreads();

  for (int it = 0; it < ntiles; ++it) {
    const int cur = it & 1, nxt = cur ^ 1;
    const bool more = (it + 1) < ntiles;

    uint4 a0 = {0,0,0,0}, a1 = {0,0,0,0}, b0 = {0,0,0,0}, b1 = {0,0,0,0};
    if (more) {
      const int ko = (it + 1) * Bb_K;
      if (aIn) { const uint4* p = (const uint4*)(Aptr + ko); a0 = p[0]; a1 = p[1]; }
      if (bIn) { const uint4* p = (const uint4*)(Bptr + ko); b0 = p[0]; b1 = p[1]; }
    }

    FragU af[2], bfr[4];
#pragma unroll
    for (int mt = 0; mt < 2; ++mt) {
      int row = wm * 32 + mt * 16 + (lane & 15);
      af[mt].q[0] = *(const uint4*)(&As[cur][row * LDT + kb]);
      af[mt].q[1] = *(const uint4*)(&As[cur][row * LDT + kb + 16]);
    }
#pragma unroll
    for (int nt = 0; nt < 4; ++nt) {
      int row = wn * 64 + nt * 16 + (lane & 15);
      bfr[nt].q[0] = *(const uint4*)(&Bs[cur][row * LDT + kb]);
      bfr[nt].q[1] = *(const uint4*)(&Bs[cur][row * LDT + kb + 16]);
    }
#pragma unroll
    for (int mt = 0; mt < 2; ++mt)
#pragma unroll
      for (int nt = 0; nt < 4; ++nt)
        acc[mt][nt] = __builtin_amdgcn_wmma_f32_16x16x32_bf16(
            false, af[mt].v, false, bfr[nt].v, (short)0, acc[mt][nt], false, false);

    if (more) {  // stage next tile into the other buffer (no race: distinct buffer)
      uint4* da = (uint4*)(&As[nxt][r * LDT + h]); da[0] = a0; da[1] = a1;
      uint4* db = (uint4*)(&Bs[nxt][r * LDT + h]); db[0] = b0; db[1] = b1;
    }
    __syncthreads();
  }

  float* Cf = (float*)Cp;
  unsigned short* Cb = (unsigned short*)Cp;
  const size_t czoff = (size_t)z * sC;
#pragma unroll
  for (int mt = 0; mt < 2; ++mt) {
#pragma unroll
    for (int nt = 0; nt < 4; ++nt) {
      int gn  = n0 + wn * 64 + nt * 16 + (lane & 15);
      int gmb = m0 + wm * 32 + mt * 16 + ((lane & 16) ? 8 : 0);
#pragma unroll
      for (int rr = 0; rr < 8; ++rr) {
        int gm = gmb + rr;
        if (gm < M && gn < N) {
          float v = acc[mt][nt][rr];
          if (bias) v += bias[gn];
          if (act == 1) v = 0.5f * v * (1.0f + erff(v * 0.70710678118f));
          size_t idx = czoff + (size_t)gm * ldc + gn;
          if (cmode == 0)      Cf[idx] = v;
          else if (cmode == 1) Cb[idx] = f2bf(v);
          else                 Cf[idx] += v;
        }
      }
    }
  }
}

// ---------------- fp32 -> bf16 cast ----------------
__global__ void cast_kernel(const float* __restrict__ in,
                            unsigned short* __restrict__ out, long long n) {
  long long i = (long long)blockIdx.x * blockDim.x + threadIdx.x;
  if (i < n) out[i] = f2bf(in[i]);
}

// ---------------- LayerNorm (row) + optional fused unary classifier dot ----------------
__global__ __launch_bounds__(256) void ln_kernel(
    const float* __restrict__ Hs, int N,
    const float* __restrict__ g, const float* __restrict__ be,
    unsigned short* __restrict__ outb,
    const float* __restrict__ dW, const float* __restrict__ dB,
    float* __restrict__ dOut)
{
  const int row = blockIdx.x;
  const int tid = threadIdx.x;
  const float* h = Hs + (size_t)row * N;
  __shared__ float red[256];
  float s = 0.f, s2 = 0.f;
  for (int i = tid; i < N; i += 256) { float v = h[i]; s += v; s2 += v * v; }
  red[tid] = s; __syncthreads();
  for (int o = 128; o > 0; o >>= 1) { if (tid < o) red[tid] += red[tid + o]; __syncthreads(); }
  float mu = red[0] / (float)N; __syncthreads();
  red[tid] = s2; __syncthreads();
  for (int o = 128; o > 0; o >>= 1) { if (tid < o) red[tid] += red[tid + o]; __syncthreads(); }
  float var = red[0] / (float)N - mu * mu; __syncthreads();
  float rstd = rsqrtf(var + 1e-5f);
  unsigned short* o = outb + (size_t)row * N;
  float dacc = 0.f;
  for (int i = tid; i < N; i += 256) {
    float v = (h[i] - mu) * rstd * g[i] + be[i];
    o[i] = f2bf(v);
    if (dW) dacc += v * dW[i];
  }
  if (dW) {
    red[tid] = dacc; __syncthreads();
    for (int ofs = 128; ofs > 0; ofs >>= 1) { if (tid < ofs) red[tid] += red[tid + ofs]; __syncthreads(); }
    if (tid == 0) dOut[row] = red[0] + dB[0];
  }
}

// ---------------- banded joint scores: one wave per (d, s, b) ----------------
__global__ __launch_bounds__(32) void band_kernel(
    const unsigned short* __restrict__ q, const unsigned short* __restrict__ em,
    const float* __restrict__ s_sc, const float* __restrict__ e_sc,
    float* __restrict__ band)
{
  const int d = blockIdx.x, s = blockIdx.y, b = blockIdx.z;
  const int lane = threadIdx.x;
  const int t = s + d;
  float out = -3.0e38f;
  if (t < SS) {
    const unsigned short* qa = q  + ((size_t)b * SS + s) * FD;
    const unsigned short* eb = em + ((size_t)b * SS + t) * FD;
    float p = 0.f;
    for (int k = lane; k < FD; k += 32) p += bf2f(qa[k]) * bf2f(eb[k]);
    for (int off = 16; off > 0; off >>= 1) p += __shfl_xor(p, off, 32);
    float v = p + s_sc[b * SS + s] + e_sc[b * SS + t];     // in-band: mask==1
    out = fminf(fmaxf(v, NEGV), -NEGV);
  }
  if (lane == 0) band[((size_t)b * SS + s) * SPAN + d] = out;
}

// ---------------- per-batch top-k (radix select) + sort of flat indices ----------------
__device__ __forceinline__ unsigned fkey(float f) {
  unsigned u = __float_as_uint(f);
  return (u & 0x80000000u) ? ~u : (u | 0x80000000u);
}

__global__ __launch_bounds__(1024) void topk_kernel(
    const float* __restrict__ band, const int* __restrict__ masks,
    int* __restrict__ starts, int* __restrict__ ends,
    float* __restrict__ tms, int* __restrict__ kcnt)
{
  const int b = blockIdx.x, tid = threadIdx.x;
  const int NC = SS * SPAN;
  const float* vb = band + (size_t)b * NC;
  __shared__ int hist[256];
  __shared__ int sel[1024];
  __shared__ int ksum, cntG, scalByte, scalNeed;

  if (tid == 0) ksum = 0;
  __syncthreads();
  int mloc = 0;
  for (int i = tid; i < SS; i += 1024) mloc += masks[b * SS + i];
  atomicAdd(&ksum, mloc);
  __syncthreads();
  int kc = (int)((float)ksum * 0.4f);
  if (kc > MAXK) kc = MAXK;

  unsigned prefix = 0, pmask = 0;
  int kneed = kc;
  for (int shift = 24; shift >= 0; shift -= 8) {
    if (tid < 256) hist[tid] = 0;
    __syncthreads();
    for (int i = tid; i < NC; i += 1024) {
      unsigned key = fkey(vb[i]);
      if ((key & pmask) == prefix) atomicAdd(&hist[(key >> shift) & 255], 1);
    }
    __syncthreads();
    if (tid == 0) {
      int cum = 0, byte = 0, rem = kneed;
      for (int bb = 255; bb >= 0; --bb) {
        int hc = hist[bb];
        if (cum + hc >= kneed) { byte = bb; rem = kneed - cum; break; }
        cum += hc;
      }
      scalByte = byte; scalNeed = rem;
    }
    __syncthreads();
    prefix |= ((unsigned)scalByte) << shift;
    pmask  |= 0xFFu << shift;
    kneed = scalNeed;
    __syncthreads();
  }

  if (tid == 0) cntG = 0;
  __syncthreads();
  for (int i = tid; i < NC; i += 1024) {
    unsigned key = fkey(vb[i]);
    if (key > prefix) {
      int p = atomicAdd(&cntG, 1);
      if (p < MAXK) { int s_ = i / SPAN, d_ = i % SPAN; sel[p] = s_ * SS + s_ + d_; }
    }
  }
  __syncthreads();
  if (tid == 0) {
    int p = cntG, rem = kneed;
    for (int i = 0; i < NC && rem > 0; ++i) {     // ties in index order (matches top_k)
      if (fkey(vb[i]) == prefix) { int s_ = i / SPAN, d_ = i % SPAN; sel[p++] = s_ * SS + s_ + d_; --rem; }
    }
    for (int qd = kc; qd < 1024; ++qd) sel[qd] = (qd < MAXK) ? (SS * SS - 1) : 0x7FFFFFFF;
  }
  __syncthreads();

  // bitonic sort ascending (1024)
  for (int ksz = 2; ksz <= 1024; ksz <<= 1) {
    for (int j = ksz >> 1; j > 0; j >>= 1) {
      int i = tid, ixj = i ^ j;
      if (ixj > i) {
        bool up = ((i & ksz) == 0);
        int a = sel[i], c = sel[ixj];
        if ((a > c) == up) { sel[i] = c; sel[ixj] = a; }
      }
      __syncthreads();
    }
  }

  if (tid < MAXK) {
    int f = sel[tid];
    int st = f / SS, en = f % SS;
    starts[b * MAXK + tid] = st;
    ends[b * MAXK + tid] = en;
    int d = en - st;
    float v = (d >= 0 && d < SPAN) ? vb[st * SPAN + d] : NEGV;
    tms[b * MAXK + tid] = v;
  }
  if (tid == 0) kcnt[b] = kc;
}

// ---------------- gather span boundary reps ----------------
__global__ __launch_bounds__(256) void gather_kernel(
    const unsigned short* __restrict__ sc, const unsigned short* __restrict__ ec,
    const int* __restrict__ starts, const int* __restrict__ ends,
    unsigned short* __restrict__ tks, unsigned short* __restrict__ tke)
{
  int slot = blockIdx.x % MAXK, b = blockIdx.x / MAXK;
  int st = starts[b * MAXK + slot], en = ends[b * MAXK + slot];
  const unsigned short* ss = sc + ((size_t)b * SS + st) * FL;
  const unsigned short* es = ec + ((size_t)b * SS + en) * FL;
  unsigned short* ds = tks + ((size_t)b * MAXK + slot) * FL;
  unsigned short* de = tke + ((size_t)b * MAXK + slot) * FL;
  for (int i = threadIdx.x; i < FL; i += 256) { ds[i] = ss[i]; de[i] = es[i]; }
}

// ---------------- final: pair + coref, mask, clamp, zero column ----------------
__global__ void final_kernel(const float* __restrict__ tms,
                             const float* __restrict__ coref,
                             const int* __restrict__ kcnt,
                             float* __restrict__ out)
{
  long long idx = (long long)blockIdx.x * blockDim.x + threadIdx.x;
  const long long total = (long long)BB * MAXK * (MAXK + 1);
  if (idx >= total) return;
  int j = (int)(idx % (MAXK + 1));
  long long r = idx / (MAXK + 1);
  int i = (int)(r % MAXK);
  int b = (int)(r / MAXK);
  float res = 0.f;
  if (j < MAXK) {
    float v = tms[b * MAXK + i] + tms[b * MAXK + j]
            + coref[((size_t)b * MAXK + i) * MAXK + j];
    float m = ((j < i) && (i < kcnt[b])) ? 1.f : 0.f;
    res = fminf(fmaxf(v + (1.f - m) * NEGV, NEGV), -NEGV);
  }
  out[idx] = res;
}

// ---------------- host orchestration ----------------
extern "C" void kernel_launch(void* const* d_in, const int* in_sizes, int n_in,
                              void* d_out, int out_size, void* d_ws, size_t ws_size,
                              hipStream_t stream) {
  (void)in_sizes; (void)n_in; (void)out_size; (void)ws_size;
  const float* input_emb  = (const float*)d_in[0];
  const int*   input_mask = (const int*)d_in[1];
  const float* W[4]  = {(const float*)d_in[2],  (const float*)d_in[6],
                        (const float*)d_in[10], (const float*)d_in[14]};
  const float* bi[4] = {(const float*)d_in[3],  (const float*)d_in[7],
                        (const float*)d_in[11], (const float*)d_in[15]};
  const float* ga[4] = {(const float*)d_in[4],  (const float*)d_in[8],
                        (const float*)d_in[12], (const float*)d_in[16]};
  const float* bt[4] = {(const float*)d_in[5],  (const float*)d_in[9],
                        (const float*)d_in[13], (const float*)d_in[17]};
  const float* cls_s_W = (const float*)d_in[18];
  const float* cls_s_b = (const float*)d_in[19];
  const float* cls_e_W = (const float*)d_in[20];
  const float* cls_e_b = (const float*)d_in[21];
  const float* s2e_W   = (const float*)d_in[22];
  const float* s2e_b   = (const float*)d_in[23];
  const float* aW[4]   = {(const float*)d_in[24], (const float*)d_in[26],
                          (const float*)d_in[28], (const float*)d_in[30]};
  const float* ab[4]   = {(const float*)d_in[25], (const float*)d_in[27],
                          (const float*)d_in[29], (const float*)d_in[31]};

  // ---- workspace carve-out ----
  char* ws = (char*)d_ws;
  size_t off = 0;
  auto carve = [&](size_t bytes) { size_t o = off; off += (bytes + 255) & ~(size_t)255; return o; };
  const size_t M1 = (size_t)BB * SS;  // 4096
  unsigned short* emb_bf = (unsigned short*)(ws + carve(M1 * HH * 2));
  unsigned short* act_bf[4];
  for (int i = 0; i < 4; ++i) act_bf[i] = (unsigned short*)(ws + carve(M1 * FD * 2));
  unsigned short* q_bf = (unsigned short*)(ws + carve(M1 * FD * 2));
  float* h_tmp = (float*)(ws + carve(M1 * FD * 4));
  float* s_sc  = (float*)(ws + carve(M1 * 4));
  float* e_sc  = (float*)(ws + carve(M1 * 4));
  float* band  = (float*)(ws + carve((size_t)BB * SS * SPAN * 4));
  int*   starts= (int*)(ws + carve((size_t)BB * MAXK * 4));
  int*   ends  = (int*)(ws + carve((size_t)BB * MAXK * 4));
  float* tms   = (float*)(ws + carve((size_t)BB * MAXK * 4));
  int*   kcnt  = (int*)(ws + carve((size_t)BB * 4));
  unsigned short* tks = (unsigned short*)(ws + carve((size_t)BB * MAXK * FL * 2));
  unsigned short* tke = (unsigned short*)(ws + carve((size_t)BB * MAXK * FL * 2));
  unsigned short* u[4];
  for (int i = 0; i < 4; ++i) u[i] = (unsigned short*)(ws + carve((size_t)BB * MAXK * FL * 2));
  float* coref = (float*)(ws + carve((size_t)BB * MAXK * MAXK * 4));
  // transposed bf16 weights: [N][K] layout for the unified GEMM B path
  unsigned short* Wt[4];
  for (int i = 0; i < 4; ++i) Wt[i] = (unsigned short*)(ws + carve((size_t)HH * FD * 2));
  unsigned short* s2eT = (unsigned short*)(ws + carve((size_t)FD * FD * 2));
  unsigned short* aWT[4];
  for (int i = 0; i < 4; ++i) aWT[i] = (unsigned short*)(ws + carve((size_t)FL * FL * 2));

  // 0. transpose-cast all weights to bf16 [N][K]
  for (int p = 0; p < 4; ++p)
    tcast_kernel<<<dim3(FD / 32, HH / 32), 256, 0, stream>>>(W[p], Wt[p], HH, FD);
  tcast_kernel<<<dim3(FD / 32, FD / 32), 256, 0, stream>>>(s2e_W, s2eT, FD, FD);
  for (int p = 0; p < 4; ++p)
    tcast_kernel<<<dim3(FL / 32, FL / 32), 256, 0, stream>>>(aW[p], aWT[p], FL, FL);

  // 1. cast input to bf16
  {
    long long n = (long long)M1 * HH;
    cast_kernel<<<(unsigned)((n + 255) / 256), 256, 0, stream>>>(input_emb, emb_bf, n);
  }

  // 2. four MLPs: GEMM(+bias+GELU) -> LN(+fused cls dot for sm/em)
  dim3 gMlp(FD / Bb_N, (unsigned)((M1 + Bb_M - 1) / Bb_M), 1);
  for (int p = 0; p < 4; ++p) {
    wmma_gemm_kernel<<<gMlp, 256, 0, stream>>>(
        emb_bf, HH, 0, Wt[p], HH, 0, bi[p],
        h_tmp, FD, 0, 0, (int)M1, FD, HH, 1);
    const float* dW = (p == 0) ? cls_s_W : (p == 1) ? cls_e_W : nullptr;
    const float* dB = (p == 0) ? cls_s_b : (p == 1) ? cls_e_b : nullptr;
    float* dO = (p == 0) ? s_sc : (p == 1) ? e_sc : nullptr;
    ln_kernel<<<(unsigned)M1, 256, 0, stream>>>(
        h_tmp, FD, ga[p], bt[p], act_bf[p], dW, dB, dO);
  }

  // 3. q = sm @ s2e_W + s2e_b  (bf16 out)
  wmma_gemm_kernel<<<gMlp, 256, 0, stream>>>(
      act_bf[0], FD, 0, s2eT, FD, 0, s2e_b,
      q_bf, FD, 0, 1, (int)M1, FD, FD, 0);

  // 4. banded joint scores
  band_kernel<<<dim3(SPAN, SS, BB), 32, 0, stream>>>(q_bf, act_bf[1], s_sc, e_sc, band);

  // 5. per-batch top-k + sort
  topk_kernel<<<BB, 1024, 0, stream>>>(band, input_mask, starts, ends, tms, kcnt);

  // 6. gather boundary reps
  gather_kernel<<<BB * MAXK, 256, 0, stream>>>(act_bf[2], act_bf[3], starts, ends, tks, tke);

  // 7. u_* = tk @ a_*_W + a_*_b  (bf16 out), M = B*MAXK rows
  {
    const int Mu = BB * MAXK;
    dim3 gU(FL / Bb_N, (unsigned)((Mu + Bb_M - 1) / Bb_M), 1);
    const unsigned short* Au[4] = {tks, tke, tks, tke};   // ss, ee, se, es
    for (int i = 0; i < 4; ++i)
      wmma_gemm_kernel<<<gU, 256, 0, stream>>>(
          Au[i], FL, 0, aWT[i], FL, 0, ab[i],
          u[i], FL, 0, 1, Mu, FL, FL, 0);
  }

  // 8. coref = u_ss@tks^T + u_ee@tke^T + u_se@tke^T + u_es@tks^T  (per batch)
  {
    dim3 gC((MAXK + Bb_N - 1) / Bb_N, (MAXK + Bb_M - 1) / Bb_M, BB);
    const long long sAB = (long long)MAXK * FL;
    const long long sC  = (long long)MAXK * MAXK;
    const unsigned short* Vv[4] = {tks, tke, tke, tks};
    for (int i = 0; i < 4; ++i)
      wmma_gemm_kernel<<<gC, 256, 0, stream>>>(
          u[i], FL, sAB, Vv[i], FL, sAB, nullptr,
          coref, MAXK, sC, (i == 0) ? 0 : 2, MAXK, MAXK, FL, 0);
  }

  // 9. final assembly
  {
    long long total = (long long)BB * MAXK * (MAXK + 1);
    final_kernel<<<(unsigned)((total + 255) / 256), 256, 0, stream>>>(
        tms, coref, kcnt, (float*)d_out);
  }
}
